// DiseaseModel_mlp_85993835200659
// MI455X (gfx1250) — compile-verified, matrix operand order checked by reading, not run on hardware
//
#include <hip/hip_runtime.h>
#include <hip/hip_bf16.h>
#include <math.h>

// ---------------------------------------------------------------------------
// MI455X (gfx1250, wave32) fused GAT + FFN + decoder.
//   grid = 64 blocks (one batch graph per block), block = 256 threads = 8 waves.
//   GEMMs -> v_wmma_f32_16x16x32_f16. ALL operands staged in LDS, transposed
//   where consumed as B, zero-padded so fragment gathers are unconditional
//   16B ds_load_b128 pairs. K=34 tiles use stride 48 + a "tail" gather for
//   the second K-step (k>=48 is identically zero -> constant upper chunk).
// ---------------------------------------------------------------------------

typedef _Float16 half_t;
typedef __attribute__((ext_vector_type(16))) _Float16 v16h;
typedef __attribute__((ext_vector_type(8)))  _Float16 v8h;
typedef __attribute__((ext_vector_type(8)))  float    v8f;

#define WMMA_F16(A_, B_, C_) \
  __builtin_amdgcn_wmma_f32_16x16x32_f16(false, (A_), false, (B_), (short)0, (C_), false, false)

// Contiguous fragment load (A from row-major tile, or B from transposed tile).
// Per CDNA5 ISA 7.12.2: lane L<16 owns line (m0|n0)+L with K {k0+0..7,k0+16..23};
// lane L>=16 owns the same line with K {k0+8..15, k0+24..31}.
__device__ __forceinline__ v16h frag_ld(const half_t* base, int line, int stride,
                                        int k0, int lane) {
  const half_t* p = base + line * stride + k0 + ((lane >> 4) << 3);
  const v8h lo = *(const v8h*)p;
  const v8h hi = *(const v8h*)(p + 16);
  v16h r;
#pragma unroll
  for (int j = 0; j < 8; ++j) { r[j] = lo[j]; r[j + 8] = hi[j]; }
  return r;
}

// Tail fragment for K-step 1 of a K=34 tile stored at stride 48:
// k in [32,48) comes from the (zero-padded) tile, k in [48,64) is exactly 0.
__device__ __forceinline__ v16h frag_ld_tail48(const half_t* base, int line, int lane) {
  const half_t* p = base + line * 48 + 32 + ((lane >> 4) << 3);
  const v8h lo = *(const v8h*)p;
  v16h r;
#pragma unroll
  for (int j = 0; j < 8; ++j) { r[j] = lo[j]; r[j + 8] = (half_t)0.0f; }
  return r;
}

// --- wave32 reductions ----------------------------------------------------
__device__ __forceinline__ float wave_max32(float v) {
#pragma unroll
  for (int off = 16; off > 0; off >>= 1) v = fmaxf(v, __shfl_xor(v, off, 32));
  return v;
}
__device__ __forceinline__ float wave_sum32(float v) {
#pragma unroll
  for (int off = 16; off > 0; off >>= 1) v += __shfl_xor(v, off, 32);
  return v;
}

// ===========================================================================
// Kernel 1: one GAT round.  xout = xin + softmax_mask(leaky(f_src+f_dst)) @ h,
//           h = relu(xin @ Wg + bg).
// ===========================================================================
__global__ __launch_bounds__(256)
void gat_layer_kernel(const float* __restrict__ xin, float* __restrict__ xout,
                      const int* __restrict__ Amat,
                      const float* __restrict__ Wg, const float* __restrict__ bg,
                      const float* __restrict__ av) {
  __shared__ __align__(16) half_t sX[128 * 48];    // x, stride 48        12 KB
  __shared__ __align__(16) half_t sHt[48 * 128];   // h^T [n][p]          12 KB
  __shared__ __align__(16) half_t sAtt[128 * 128]; // attention           32 KB
  __shared__ __align__(16) half_t sWgT[48 * 48];   // Wg^T [n][k]        4.5 KB
  __shared__ float sFs[128];
  __shared__ float sFd[128];

  const int b    = blockIdx.x;
  const int tid  = threadIdx.x;
  const int lane = tid & 31;
  const int wave = tid >> 5;
  const float* xb = xin  + (size_t)b * (128 * 34);
  const int*   Ab = Amat + (size_t)b * (128 * 128);

  __builtin_prefetch(Ab, 0, 1);   // global_prefetch_b8: adjacency toward GL2

  // stage x -> f16 (data + pad loops, both unconditional)
  for (int i = tid; i < 128 * 34; i += 256) {
    const int r = i / 34, c = i - r * 34;
    sX[r * 48 + c] = (half_t)xb[i];
  }
  for (int i = tid; i < 128 * 14; i += 256) {
    const int r = i / 14, c = 34 + (i - r * 14);
    sX[r * 48 + c] = (half_t)0.0f;
  }
  // stage Wg^T with zero padding (clamped loads + register selects)
  for (int i = tid; i < 48 * 48; i += 256) {
    const int n = i / 48, k = i - n * 48;
    const float w = Wg[((k < 34) ? k : 33) * 34 + ((n < 34) ? n : 33)];
    sWgT[i] = (n < 34 && k < 34) ? (half_t)w : (half_t)0.0f;
  }
  // zero sHt's pad lines (n = 34..47) once
  for (int i = 34 * 128 + tid; i < 48 * 128; i += 256) sHt[i] = (half_t)0.0f;
  __syncthreads();

  const int m0 = wave * 16;

  // ---- h = relu(x @ Wg + bg) : M=128, N=34(->48), K=34(->64) ----
#pragma unroll
  for (int nt = 0; nt < 3; ++nt) {
    const int n0 = nt * 16;
    v8f c = {};
    {
      v16h a  = frag_ld(sX,   m0 + (lane & 15), 48, 0, lane);
      v16h bm = frag_ld(sWgT, n0 + (lane & 15), 48, 0, lane);
      c = WMMA_F16(a, bm, c);
      a  = frag_ld_tail48(sX,   m0 + (lane & 15), lane);
      bm = frag_ld_tail48(sWgT, n0 + (lane & 15), lane);
      c = WMMA_F16(a, bm, c);
    }
    const int col   = n0 + (lane & 15);
    const int rbase = m0 + ((lane >> 4) << 3);
    const float bias = bg[(col < 34) ? col : 33];
    v8h hv;
#pragma unroll
    for (int r = 0; r < 8; ++r) hv[r] = (half_t)fmaxf(c[r] + bias, 0.0f);
    if (col < 34)                          // one branch per fragment, b128 store
      *(v8h*)&sHt[col * 128 + rbase] = hv;
  }
  __syncthreads();

  // ---- attention logits per node: f = h . av  (h read from transposed tile)
  if (tid < 128) {
    float s = 0.f, d = 0.f;
    for (int k = 0; k < 34; ++k) {
      const float hv = (float)sHt[k * 128 + tid];
      s += hv * av[k];
      d += hv * av[34 + k];
    }
    sFs[tid] = s;
    sFd[tid] = d;
  }
  __syncthreads();

  // ---- masked row softmax (wave per row, lane covers 4 columns) ----
  for (int it = 0; it < 16; ++it) {
    const int row = wave * 16 + it;
    const float fs = sFs[row];
    float e[4];
    float mx = -3.4e38f;
#pragma unroll
    for (int j = 0; j < 4; ++j) {
      const int col = lane + 32 * j;
      float v = fs + sFd[col];
      v = (v > 0.f) ? v : 0.01f * v;        // leaky_relu(0.01)
      v = (Ab[row * 128 + col] > 0) ? v : -9.0e15f;
      e[j] = v;
      mx = fmaxf(mx, v);
    }
    mx = wave_max32(mx);
    float sum = 0.f;
#pragma unroll
    for (int j = 0; j < 4; ++j) {
      e[j] = __expf(e[j] - mx);             // fully-masked row -> uniform, as in ref
      sum += e[j];
    }
    sum = wave_sum32(sum);
    const float inv = 1.0f / sum;
#pragma unroll
    for (int j = 0; j < 4; ++j)
      sAtt[row * 128 + lane + 32 * j] = (half_t)(e[j] * inv);
  }
  __syncthreads();

  // ---- xout = xin + att @ h : M=128, N=34(->48), K=128 (all-contiguous) ----
  float* xo = xout + (size_t)b * (128 * 34);
#pragma unroll
  for (int nt = 0; nt < 3; ++nt) {
    const int n0 = nt * 16;
    v8f c = {};
#pragma unroll
    for (int ks = 0; ks < 4; ++ks) {
      v16h a  = frag_ld(sAtt, m0 + (lane & 15), 128, ks * 32, lane);  // A: att
      v16h bm = frag_ld(sHt,  n0 + (lane & 15), 128, ks * 32, lane);  // B: h^T
      c = WMMA_F16(a, bm, c);
    }
    const int col   = n0 + (lane & 15);
    const int rbase = m0 + ((lane >> 4) << 3);
    if (col < 34) {
#pragma unroll
      for (int r = 0; r < 8; ++r) {
        const int row = rbase + r;
        xo[row * 34 + col] = xb[row * 34 + col] + c[r];
      }
    }
  }
}

// ===========================================================================
// Kernel 2: g = relu(x@Wt+bt); d = g@Wf + bf + xs; d2 = d@Wf2 + bf2;
//           latent = max over atoms; vec = sigmoid([latent, cell_emb]).
// HID=256 in 4 tiles of 64; Wt^T and Wf^T tiles staged in LDS per iteration
// (shared by all 8 waves); afterwards sX is rewritten with d (f16) and sG
// with Wf2^T for the final GEMM.
// ===========================================================================
__global__ __launch_bounds__(256)
void ffn_kernel(const float* __restrict__ xin, const float* __restrict__ xs0,
                const float* __restrict__ cell,
                const float* __restrict__ Wt, const float* __restrict__ bt,
                const float* __restrict__ Wf, const float* __restrict__ bf,
                const float* __restrict__ Wf2, const float* __restrict__ bf2,
                float* __restrict__ vec) {
  __shared__ __align__(16) half_t sX[128 * 48];    // x (later: d)       12 KB
  __shared__ __align__(16) float  sD[128 * 34];    // d accum (f32)      17 KB
  __shared__ __align__(16) half_t sG[128 * 64];    // g tile / Wf2^T     16 KB
  __shared__ __align__(16) half_t sWtT[64 * 48];   // Wt tile^T           6 KB
  __shared__ __align__(16) half_t sWfT[48 * 64];   // Wf tile^T           6 KB
  __shared__ float sLat[8 * 128];                  //                     4 KB

  const int b    = blockIdx.x;
  const int tid  = threadIdx.x;
  const int lane = tid & 31;
  const int wave = tid >> 5;
  const float* xb  = xin + (size_t)b * (128 * 34);
  const float* x0b = xs0 + (size_t)b * (128 * 34);

  for (int i = tid; i < 128 * 34; i += 256) {
    const int r = i / 34, c = i - r * 34;
    sX[r * 48 + c] = (half_t)xb[i];
    sD[i] = x0b[i] + bf[c];                 // residual h0 + bias, accumulated below
  }
  for (int i = tid; i < 128 * 14; i += 256) {
    const int r = i / 14, c = 34 + (i - r * 14);
    sX[r * 48 + c] = (half_t)0.0f;
  }
  // one-time zeroing of pad regions (never overwritten by staging loops)
  for (int i = tid; i < 64 * 14; i += 256) {          // sWtT: k = 34..47
    const int n = i / 14, k = 34 + (i - n * 14);
    sWtT[n * 48 + k] = (half_t)0.0f;
  }
  for (int i = 34 * 64 + tid; i < 48 * 64; i += 256)  // sWfT: n = 34..47
    sWfT[i] = (half_t)0.0f;
  __syncthreads();

  const int m0 = wave * 16;

  for (int t = 0; t < 4; ++t) {             // 4 HID tiles of 64
    const int hb = t * 64;
    // stage Wt[:, hb:hb+64]^T and Wf[hb:hb+64, :34]^T (data regions only)
    for (int i = tid; i < 64 * 34; i += 256) {
      const int n = i / 34, k = i - n * 34;
      sWtT[n * 48 + k] = (half_t)Wt[k * 256 + hb + n];
    }
    for (int i = tid; i < 34 * 64; i += 256) {
      const int n = i >> 6, k = i & 63;
      sWfT[n * 64 + k] = (half_t)Wf[(hb + k) * 34 + n];
    }
    __syncthreads();

    // g_tile = relu(x @ Wt_tile + bt)  -- all-contiguous operands
#pragma unroll
    for (int nt = 0; nt < 4; ++nt) {
      const int n0 = nt * 16;
      v8f c = {};
      {
        v16h a  = frag_ld(sX,   m0 + (lane & 15), 48, 0, lane);
        v16h bm = frag_ld(sWtT, n0 + (lane & 15), 48, 0, lane);
        c = WMMA_F16(a, bm, c);
        a  = frag_ld_tail48(sX,   m0 + (lane & 15), lane);
        bm = frag_ld_tail48(sWtT, n0 + (lane & 15), lane);
        c = WMMA_F16(a, bm, c);
      }
      const int col   = n0 + (lane & 15);
      const int rbase = m0 + ((lane >> 4) << 3);
      const float bias = bt[hb + col];
#pragma unroll
      for (int r = 0; r < 8; ++r)
        sG[(rbase + r) * 64 + col] = (half_t)fmaxf(c[r] + bias, 0.0f);
    }
    __syncthreads();

    // d += g_tile @ Wf_tile  (C preloaded from sD -> accumulate)
#pragma unroll
    for (int nt = 0; nt < 3; ++nt) {
      const int n0    = nt * 16;
      const int col   = n0 + (lane & 15);
      const int rbase = m0 + ((lane >> 4) << 3);
      v8f c = {};
      if (col < 34) {
#pragma unroll
        for (int r = 0; r < 8; ++r) c[r] = sD[(rbase + r) * 34 + col];
      }
#pragma unroll
      for (int ks = 0; ks < 2; ++ks) {
        v16h a  = frag_ld(sG,   m0 + (lane & 15), 64, ks * 32, lane);
        v16h bm = frag_ld(sWfT, n0 + (lane & 15), 64, ks * 32, lane);
        c = WMMA_F16(a, bm, c);
      }
      if (col < 34) {
#pragma unroll
        for (int r = 0; r < 8; ++r) sD[(rbase + r) * 34 + col] = c[r];
      }
    }
    __syncthreads();                        // before next tile restages sWtT/sWfT
  }

  // repurpose sX <- d (f16; pad cols stay zero), sG <- Wf2^T (K-padded to 64)
  for (int i = tid; i < 128 * 34; i += 256) {
    const int r = i / 34, c = i - r * 34;
    sX[r * 48 + c] = (half_t)sD[i];
  }
  for (int i = tid; i < 128 * 64; i += 256) {
    const int n = i >> 6, k = i & 63;
    const float w = Wf2[((k < 34) ? k : 33) * 128 + n];   // clamped load
    sG[i] = (k < 34) ? (half_t)w : (half_t)0.0f;          // register select
  }
  __syncthreads();

  // d2 = d @ Wf2 + bf2 ; per-tile max over 16 rows, combined across waves.
#pragma unroll
  for (int nt = 0; nt < 8; ++nt) {
    const int n0 = nt * 16;
    v8f c = {};
    {
      v16h a  = frag_ld(sX, m0 + (lane & 15), 48, 0, lane);
      v16h bm = frag_ld(sG, n0 + (lane & 15), 64, 0, lane);
      c = WMMA_F16(a, bm, c);
      a  = frag_ld_tail48(sX, m0 + (lane & 15), lane);
      bm = frag_ld(sG, n0 + (lane & 15), 64, 32, lane);
      c = WMMA_F16(a, bm, c);
    }
    const int col  = n0 + (lane & 15);
    const float bias = bf2[col];
    float m = -3.4e38f;
#pragma unroll
    for (int r = 0; r < 8; ++r) m = fmaxf(m, c[r] + bias);
    m = fmaxf(m, __shfl_xor(m, 16, 32));    // fold the two 8-row halves
    sLat[wave * 128 + col] = m;             // lanes l and l+16 write same value
  }
  __syncthreads();

  float* vb = vec + (size_t)b * 640;
  if (tid < 128) {
    float m = sLat[tid];
#pragma unroll
    for (int w = 1; w < 8; ++w) m = fmaxf(m, sLat[w * 128 + tid]);
    vb[tid] = 1.0f / (1.0f + __expf(-m));
  }
  for (int i = tid; i < 512; i += 256) {
    const float v = cell[(size_t)b * 512 + i];
    vb[128 + i] = 1.0f / (1.0f + __expf(-v));
  }
}

// ===========================================================================
// Kernel 3: decoder MLP 640->128->256->512->1 (VALU; tiny per batch).
// ===========================================================================
__global__ __launch_bounds__(256)
void decoder_kernel(const float* __restrict__ vec,
                    const float* __restrict__ W1, const float* __restrict__ b1,
                    const float* __restrict__ W2, const float* __restrict__ b2,
                    const float* __restrict__ W3, const float* __restrict__ b3,
                    const float* __restrict__ W4, const float* __restrict__ b4,
                    float* __restrict__ out) {
  __shared__ float sV[640];
  __shared__ float sH1[128];
  __shared__ float sH2[256];
  __shared__ float sH3[512];
  __shared__ float sRed[8];

  const int b    = blockIdx.x;
  const int tid  = threadIdx.x;
  const int lane = tid & 31;
  const int wave = tid >> 5;

  for (int i = tid; i < 640; i += 256) sV[i] = vec[(size_t)b * 640 + i];
  __syncthreads();

  if (tid < 128) {
    float s = b1[tid];
    for (int k = 0; k < 640; ++k) s += sV[k] * W1[k * 128 + tid];
    sH1[tid] = fmaxf(s, 0.0f);
  }
  __syncthreads();

  {
    float s = b2[tid];
    for (int k = 0; k < 128; ++k) s += sH1[k] * W2[k * 256 + tid];
    sH2[tid] = fmaxf(s, 0.0f);
  }
  __syncthreads();

  for (int o = tid; o < 512; o += 256) {
    float s = b3[o];
    for (int k = 0; k < 256; ++k) s += sH2[k] * W3[k * 512 + o];
    sH3[o] = fmaxf(s, 0.0f);
  }
  __syncthreads();

  float p = 0.0f;
  for (int i = tid; i < 512; i += 256) p += sH3[i] * W4[i];
  p = wave_sum32(p);
  if (lane == 0) sRed[wave] = p;
  __syncthreads();
  if (tid == 0) {
    float s = b4[0];
#pragma unroll
    for (int w = 0; w < 8; ++w) s += sRed[w];
    out[b] = s;
  }
}

// ===========================================================================
extern "C" void kernel_launch(void* const* d_in, const int* in_sizes, int n_in,
                              void* d_out, int out_size, void* d_ws, size_t ws_size,
                              hipStream_t stream) {
  (void)in_sizes; (void)n_in; (void)out_size; (void)ws_size;

  const float* xs   = (const float*)d_in[0];
  const int*   Am   = (const int*)d_in[1];
  const float* cell = (const float*)d_in[2];
  const float* Wg   = (const float*)d_in[3];   // [3,34,34]
  const float* bg   = (const float*)d_in[4];   // [3,34]
  const float* av   = (const float*)d_in[5];   // [3,68]
  const float* Wt   = (const float*)d_in[6];
  const float* bt   = (const float*)d_in[7];
  const float* Wf   = (const float*)d_in[8];
  const float* bf   = (const float*)d_in[9];
  const float* Wf2  = (const float*)d_in[10];
  const float* bf2  = (const float*)d_in[11];
  const float* W1   = (const float*)d_in[12];
  const float* b1   = (const float*)d_in[13];
  const float* W2   = (const float*)d_in[14];
  const float* b2   = (const float*)d_in[15];
  const float* W3   = (const float*)d_in[16];
  const float* b3   = (const float*)d_in[17];
  const float* W4   = (const float*)d_in[18];
  const float* b4   = (const float*)d_in[19];

  const size_t XN = (size_t)64 * 128 * 34;     // x ping-pong buffers in ws
  float* wx1  = (float*)d_ws;
  float* wx2  = wx1 + XN;
  float* wvec = wx2 + XN;                      // [64,640]  (total ws ~2.3 MB)

  dim3 grid(64), block(256);
  gat_layer_kernel<<<grid, block, 0, stream>>>(xs,  wx1, Am, Wg,        bg,      av);
  gat_layer_kernel<<<grid, block, 0, stream>>>(wx1, wx2, Am, Wg + 1156, bg + 34, av + 68);
  gat_layer_kernel<<<grid, block, 0, stream>>>(wx2, wx1, Am, Wg + 2312, bg + 68, av + 136);
  ffn_kernel<<<grid, block, 0, stream>>>(wx1, xs, cell, Wt, bt, Wf, bf, Wf2, bf2, wvec);
  decoder_kernel<<<grid, block, 0, stream>>>(wvec, W1, b1, W2, b2, W3, b3, W4, b4,
                                             (float*)d_out);
}
